// Model_53996328845741
// MI455X (gfx1250) — matrix operand order repaired and checked by reading, not compiled, not run on hardware
//
#include <hip/hip_runtime.h>
#include <math.h>

typedef float  v2f   __attribute__((ext_vector_type(2)));
typedef float  v8f   __attribute__((ext_vector_type(8)));
typedef __bf16 v16bf __attribute__((ext_vector_type(16)));

union V8F   { v8f v; float f[8]; };
union V16BF { v16bf v; uint4 q[2]; };

#define NB   8
#define NL   1024
#define DMOD 512
#define NH   8
#define DHD  64
#define KPAD 68      /* 33 complex bins -> 66 basis rows, padded to 68 (K%4==0) */
#define TWO_PI 6.283185307179586f

__device__ __forceinline__ unsigned int f2bf(float f) {
  unsigned int u = __float_as_uint(f);
  return (u + 0x7FFFu + ((u >> 16) & 1u)) >> 16;   // round-to-nearest-even bf16
}

// ---------------- embedding: out[b,l,d] = sum_c x[b,l,c]*w[c,d] + bias[d] ----------------
__global__ void embed_kernel(const float* __restrict__ x, const float* __restrict__ w,
                             const float* __restrict__ bias, float* __restrict__ out) {
  int idx = blockIdx.x * blockDim.x + threadIdx.x;
  if (idx >= NB*NL*DMOD) return;
  int d  = idx & (DMOD-1);
  int bl = idx >> 9;
  float acc = bias[d];
  #pragma unroll
  for (int cc = 0; cc < 7; ++cc) acc += x[bl*7 + cc] * w[cc*DMOD + d];
  out[idx] = acc;
}

// ---------------- series_decomp: 25-tap avg (count_include_pad) ----------------
__global__ void decomp_kernel(const float* __restrict__ x, float* __restrict__ seas,
                              float* __restrict__ trend) {
  int idx = blockIdx.x * blockDim.x + threadIdx.x;
  if (idx >= NB*NL*DMOD) return;
  int d = idx & (DMOD-1);
  int l = (idx >> 9) & (NL-1);
  int b = idx >> 19;
  const float* col = x + (size_t)b*NL*DMOD + d;
  int lo = l-12 < 0 ? 0 : l-12;
  int hi = l+12 > NL-1 ? NL-1 : l+12;
  float s = 0.f;
  for (int t = lo; t <= hi; ++t) s += col[(size_t)t*DMOD];
  float tr = s * (1.0f/25.0f);
  trend[idx] = tr;
  seas[idx]  = x[idx] - tr;
}

// ---------------- basis[p][j]: p=2m -> cos(2pi jm/1024), p=2m+1 -> sin ----------------
__global__ void basis_kernel(float* __restrict__ basis) {
  int idx = blockIdx.x * blockDim.x + threadIdx.x;
  if (idx >= KPAD*1024) return;
  int p = idx >> 10, j = idx & 1023;
  float v = 0.f;
  if (p < 66) {
    int m = p >> 1;
    int n = (j * m) & 1023;               // exact periodic reduction
    float sn, cn; __sincosf(TWO_PI * (float)n / 1024.0f, &sn, &cn);
    v = (p & 1) ? sn : cn;
  }
  basis[idx] = v;
}

// ---------------- coeff: per (b,i) all heads: P = Fq*conj(Fk), minus head-mean ----------------
__global__ void coeff_kernel(const float* __restrict__ q, const float* __restrict__ k,
                             float* __restrict__ coeff) {
  __shared__ float ct[64], st[64];
  __shared__ float qs[512], ks_[512];
  __shared__ float Ps[NH*33*2];
  __shared__ float Ms[33*2];
  const int tid = threadIdx.x;
  const int i = blockIdx.x & (NL-1);
  const int b = blockIdx.x >> 10;
  if (tid < 64) {
    float s, c; __sincosf(TWO_PI * (float)tid / 64.0f, &s, &c);
    ct[tid] = c; st[tid] = s;
  }
  const size_t row = ((size_t)b*NL + i) * DMOD;
  for (int e = tid; e < 512; e += blockDim.x) { qs[e] = q[row+e]; ks_[e] = k[row+e]; }
  __syncthreads();
  if (tid < NH*33) {
    int h = tid / 33, m = tid % 33;
    const float* qh = qs  + h*64;
    const float* kh = ks_ + h*64;
    float qr=0.f, qi=0.f, kr=0.f, ki=0.f;
    for (int t = 0; t < 64; ++t) {
      int n = (t*m) & 63; float c = ct[n], s = st[n];
      qr += qh[t]*c; qi -= qh[t]*s;
      kr += kh[t]*c; ki -= kh[t]*s;
    }
    Ps[tid*2+0] = qr*kr + qi*ki;       // Re(Fq*conj(Fk))
    Ps[tid*2+1] = qi*kr - qr*ki;       // Im
  }
  __syncthreads();
  if (tid < 33) {
    float mr=0.f, mi=0.f;
    for (int h = 0; h < NH; ++h) { mr += Ps[(h*33+tid)*2]; mi += Ps[(h*33+tid)*2+1]; }
    Ms[tid*2] = mr*0.125f; Ms[tid*2+1] = mi*0.125f;
  }
  __syncthreads();
  if (tid < NH*33) {
    int h = tid / 33, m = tid % 33;
    float sc = (m == 0) ? (1.0f/1024.0f) : (2.0f/1024.0f);   // irfft scale folded in
    float pr = (Ps[tid*2]   - Ms[m*2])   * sc;
    float pi = (Ps[tid*2+1] - Ms[m*2+1]) * sc;
    size_t r = (((size_t)(b*NH + h))*NL + i) * KPAD;
    coeff[r + 2*m]     = pr;           // * cos
    coeff[r + 2*m + 1] = -pi;          // * sin  (Re(P e^{i t}) = Pr cos - Pi sin)
  } else if (tid < NH*33 + NH) {
    int h = tid - NH*33;
    size_t r = (((size_t)(b*NH + h))*NL + i) * KPAD;
    coeff[r + 66] = 0.f; coeff[r + 67] = 0.f;
  }
}

// ---------------- vt: V [B,L,512] f32 -> Vt bf16 [bh][di=64][s=1024] (transposed) ----------------
__global__ void vt_kernel(const float* __restrict__ v, unsigned short* __restrict__ vt) {
  __shared__ float ts[32*65];
  const int bh = blockIdx.x >> 5;
  const int s0 = (blockIdx.x & 31) << 5;
  const int b = bh >> 3, h = bh & 7;
  const int tid = threadIdx.x;
  for (int e = tid; e < 32*64; e += 256) {
    int r = e >> 6, di = e & 63;
    ts[r*65 + di] = v[(((size_t)b*NL) + s0 + r)*DMOD + h*64 + di];
  }
  __syncthreads();
  int di = tid >> 2, ch = tid & 3;
  unsigned int p[4];
  #pragma unroll
  for (int g = 0; g < 4; ++g) {
    int sl = ch*8 + g*2;
    p[g] = f2bf(ts[sl*65 + di]) | (f2bf(ts[(sl+1)*65 + di]) << 16);
  }
  *(uint4*)(vt + (size_t)bh*DHD*NL + (size_t)di*NL + s0 + ch*8) = make_uint4(p[0],p[1],p[2],p[3]);
}

// ------- corr = coeff @ basis (f32 WMMA), softmax over lags, transposed bf16 W out -------
__global__ void __launch_bounds__(256) corrsm_kernel(const float* __restrict__ coeff,
                                                     const float* __restrict__ basis,
                                                     unsigned short* __restrict__ wt) {
  __shared__ float cs[16*1024];                       // 64 KB corr tile [s_local][j]
  const int bh = blockIdx.x >> 6;
  const int s0 = (blockIdx.x & 63) << 4;
  const int tid = threadIdx.x;
  const int wv = tid >> 5, lane = tid & 31;
  const int hlf = lane >> 4, l15 = lane & 15;
  const float* arow = coeff + (((size_t)bh*NL) + s0 + l15)*KPAD + hlf*2;  // A: M=l15, K pair
  for (int jj = 0; jj < 8; ++jj) {
    const int j = ((wv*8 + jj) << 4) + l15;           // B/C column (N)
    v8f c = {};
    #pragma unroll
    for (int ks = 0; ks < KPAD; ks += 4) {
      v2f a = *(const v2f*)(arow + ks);
      v2f b;
      b.x = basis[(ks + hlf*2    )*1024 + j];
      b.y = basis[(ks + hlf*2 + 1)*1024 + j];
      c = __builtin_amdgcn_wmma_f32_16x16x4_f32(false, a, false, b, (short)0, c, false, false);
    }
    V8F cu; cu.v = c;
    #pragma unroll
    for (int g = 0; g < 8; ++g) cs[(g + hlf*8)*1024 + j] = cu.f[g];
  }
  __syncthreads();
  // softmax: wave wv owns rows 2wv, 2wv+1 (1024 lags each)
  for (int rr = 0; rr < 2; ++rr) {
    float* rowp = cs + (wv*2 + rr)*1024;
    float m = -3.4e38f;
    for (int j = lane; j < 1024; j += 32) m = fmaxf(m, rowp[j]);
    for (int off = 16; off; off >>= 1) m = fmaxf(m, __shfl_xor(m, off, 32));
    float sum = 0.f;
    for (int j = lane; j < 1024; j += 32) { float e = __expf(rowp[j]-m); rowp[j] = e; sum += e; }
    for (int off = 16; off; off >>= 1) sum += __shfl_xor(sum, off, 32);
    float inv = 1.0f / sum;
    for (int j = lane; j < 1024; j += 32) rowp[j] *= inv;
  }
  __syncthreads();
  // write transposed bf16: wt[bh][a][s0..s0+15]  (so the GEMM's B loads are contiguous b128)
  unsigned short* wrow = wt + (size_t)bh*NL*NL + s0;
  for (int a = tid; a < 1024; a += 256) {
    unsigned int p[8];
    #pragma unroll
    for (int g = 0; g < 8; ++g)
      p[g] = f2bf(cs[(2*g)*1024 + a]) | (f2bf(cs[(2*g+1)*1024 + a]) << 16);
    *(uint4*)(wrow + (size_t)a*1024)     = make_uint4(p[0],p[1],p[2],p[3]);
    *(uint4*)(wrow + (size_t)a*1024 + 8) = make_uint4(p[4],p[5],p[6],p[7]);
  }
}

// ------- V-GEMM: D'[di,a] = sum_s Vt[di,s]*Wt[a,s]  (bf16 WMMA, f32 acc), scrambled store -------
// Each wave owns one 16-row di tile and TWO 16-col a tiles: two independent WMMA
// accumulator chains (breaks the matrix-pipe RAW chain) sharing one A stream.
__global__ void __launch_bounds__(128) vgemm_kernel(const unsigned short* __restrict__ vt,
                                                    const unsigned short* __restrict__ wt,
                                                    float* __restrict__ out) {
  const int bh = blockIdx.x >> 5;
  const int a0 = (blockIdx.x & 31) << 5;               // 32 a-columns per block
  const int b = bh >> 3, h = bh & 7;
  const int wv = threadIdx.x >> 5, lane = threadIdx.x & 31;
  const int hlf = lane >> 4, l15 = lane & 15;
  const int di0 = wv << 4;
  const unsigned short* ap  = vt + (size_t)bh*DHD*NL + (size_t)(di0 + l15)*NL;      // A row M=di
  const unsigned short* bp0 = wt + (size_t)bh*NL*NL  + (size_t)(a0  + l15)*NL;      // B col N=a
  const unsigned short* bp1 = bp0 + (size_t)16*NL;                                  // B col N=a+16
  v8f c0 = {}, c1 = {};
  for (int s0 = 0; s0 < NL; s0 += 32) {
    if (s0 + 32 < NL) {                                // prefetch next K-chunk of W stream
      __builtin_prefetch(bp0 + s0 + 32 + hlf*16, 0, 1);
      __builtin_prefetch(bp1 + s0 + 32 + hlf*16, 0, 1);
    }
    V16BF A, B0, B1;
    A.q[0]  = *(const uint4*)(ap  + s0 + hlf*16);      // A: K 0..7  / 16..23
    A.q[1]  = *(const uint4*)(ap  + s0 + hlf*16 + 8);  //    K 8..15 / 24..31
    B0.q[0] = *(const uint4*)(bp0 + s0 + hlf*16);      // B: K 0..7  / 16..23
    B0.q[1] = *(const uint4*)(bp0 + s0 + hlf*16 + 8);  //    K 8..15 / 24..31
    B1.q[0] = *(const uint4*)(bp1 + s0 + hlf*16);
    B1.q[1] = *(const uint4*)(bp1 + s0 + hlf*16 + 8);
    c0 = __builtin_amdgcn_wmma_f32_16x16x32_bf16(false, A.v, false, B0.v, (short)0, c0, false, false);
    c1 = __builtin_amdgcn_wmma_f32_16x16x32_bf16(false, A.v, false, B1.v, (short)0, c1, false, false);
  }
  V8F u0, u1; u0.v = c0; u1.v = c1;
  // out[b][l',e'] with flat = di*(H*L) + h*L + a  (faithful permute+reshape reinterpretation)
  float* op = out + (size_t)b*(NL*DMOD) + (size_t)h*NL;
  #pragma unroll
  for (int g = 0; g < 8; ++g) {
    int M = di0 + g + hlf*8;
    op[(size_t)M*(NH*NL) + a0 + l15]      = u0.f[g];   // lanes -> consecutive a: coalesced
    op[(size_t)M*(NH*NL) + a0 + 16 + l15] = u1.f[g];
  }
}

// ---------------- final: (seasonal + dec_trend + enc_trend) @ proj_w + proj_b ----------------
__global__ void final_kernel(const float* __restrict__ at, const float* __restrict__ dt,
                             const float* __restrict__ et, const float* __restrict__ pw,
                             const float* __restrict__ pb, float* __restrict__ out) {
  int idx = blockIdx.x * blockDim.x + threadIdx.x;
  if (idx >= NB*NL*7) return;
  int c  = idx % 7;
  int bl = idx / 7;
  const float* a  = at + (size_t)bl*DMOD;
  const float* d1 = dt + (size_t)bl*DMOD;
  const float* d2 = et + (size_t)bl*DMOD;
  float acc = pb[c];
  for (int d = 0; d < DMOD; ++d) acc += (a[d] + d1[d] + d2[d]) * pw[d*7 + c];
  out[idx] = acc;
}

extern "C" void kernel_launch(void* const* d_in, const int* in_sizes, int n_in,
                              void* d_out, int out_size, void* d_ws, size_t ws_size,
                              hipStream_t stream) {
  const float* x_enc  = (const float*)d_in[0];
  const float* x_dec  = (const float*)d_in[2];
  const float* enc_w  = (const float*)d_in[4];
  const float* enc_b  = (const float*)d_in[5];
  const float* dec_w  = (const float*)d_in[6];
  const float* dec_b  = (const float*)d_in[7];
  const float* proj_w = (const float*)d_in[8];
  const float* proj_b = (const float*)d_in[9];
  float* out = (float*)d_out;

  const size_t N4 = (size_t)NB*NL*DMOD;          // 4,194,304 floats
  float* ws   = (float*)d_ws;
  float* ENCE = ws + 0*N4;
  float* DECE = ws + 1*N4;
  float* ESA  = ws + 2*N4;   // encoder seasonal ping
  float* ESB  = ws + 3*N4;   // encoder seasonal pong
  float* ETR  = ws + 4*N4;   // encoder trend
  float* DSE  = ws + 5*N4;   // decoder seasonal
  float* DTR  = ws + 6*N4;   // decoder trend
  float* DAT  = ws + 7*N4;   // decoder attention output
  float* COEF = ws + 8*N4;
  const size_t COEFSZ = (size_t)NB*NH*NL*KPAD;   // 4,456,448 floats
  float* BAS  = COEF + COEFSZ;
  const size_t BASSZ = (size_t)KPAD*1024;
  unsigned short* WT = (unsigned short*)(BAS + BASSZ);   // bf16 [bh][a][s] : 128 MB
  unsigned short* VT = WT + (size_t)NB*NH*NL*NL;         // bf16 [bh][di][s]: 8 MB

  const int TPB = 256;
  const int gE = (int)((N4 + TPB-1)/TPB);
  embed_kernel <<<gE, TPB, 0, stream>>>(x_enc, enc_w, enc_b, ENCE);
  embed_kernel <<<gE, TPB, 0, stream>>>(x_dec, dec_w, dec_b, DECE);
  decomp_kernel<<<gE, TPB, 0, stream>>>(ENCE, ESA, ETR);
  decomp_kernel<<<gE, TPB, 0, stream>>>(DECE, DSE, DTR);
  basis_kernel <<<(KPAD*1024+TPB-1)/TPB, TPB, 0, stream>>>(BAS);

  auto autocorr = [&](const float* q, const float* k, const float* v, float* o) {
    coeff_kernel <<<NB*NL,    288, 0, stream>>>(q, k, COEF);
    vt_kernel    <<<NB*NH*32, 256, 0, stream>>>(v, VT);
    corrsm_kernel<<<NB*NH*64, 256, 0, stream>>>(COEF, BAS, WT);
    vgemm_kernel <<<NB*NH*32, 128, 0, stream>>>(VT, WT, o);
  };
  autocorr(ESA, ESA, ESA, ESB);      // encoder layer 1
  autocorr(ESB, ESB, ESB, ESA);      // encoder layer 2
  autocorr(DSE, ESA, ESA, DAT);      // decoder layer

  final_kernel<<<(NB*NL*7+TPB-1)/TPB, TPB, 0, stream>>>(DAT, DTR, ETR, proj_w, proj_b, out);
  (void)in_sizes; (void)n_in; (void)out_size; (void)ws_size;
}
